// MultiHeadAttention_79903571574938
// MI455X (gfx1250) — matrix operand order repaired and checked by reading, not compiled, not run on hardware
//
#include <hip/hip_runtime.h>
#include <hip/hip_bf16.h>

typedef __bf16 bf16;
typedef __attribute__((ext_vector_type(16))) __bf16 v16bf;
typedef __attribute__((ext_vector_type(8)))  float  v8f;

#define ED   1024
#define NH   16
#define HD   64
#define NTOK 2048
#define BATCH 2
#define MTOT (BATCH*NTOK)   // 4096 tokens total

// LDS byte offset (wave-relative) of a shared-memory pointer
#define LDS_OFF(p) ((unsigned)(size_t)(__attribute__((address_space(3))) const void*)(p))

__device__ __forceinline__ bf16 tobf(float f) { return (bf16)f; }

__device__ __forceinline__ v8f wmma_bf16(v16bf a, v16bf b, v8f c) {
  // D = A(16x32 bf16) * B(32x16 bf16) + C(16x16 f32)
  return __builtin_amdgcn_wmma_f32_16x16x32_bf16(false, a, false, b, (short)0, c,
                                                 false, false);
}

// Async DMA: 16 bytes per lane, global -> LDS, tracked by ASYNCcnt.
__device__ __forceinline__ void async_copy_b128(unsigned lds_off, const void* gaddr) {
  asm volatile("global_load_async_to_lds_b128 %0, %1, off"
               :: "v"(lds_off), "v"(gaddr) : "memory");
}
__device__ __forceinline__ void wait_async0() {
  asm volatile("s_wait_asynccnt 0x0" ::: "memory");
}

// -------------------------------------------------------------------------
// Projection: Y = X @ W^T + bias  (X:[4096,1024] f32, W:[out,in] f32)
// Output bf16 laid out [B, NH, N, HD]; scale folded in (1/sqrt(HD) for Q).
// Block tile 64(M) x 128(N), K-step 32; 8 waves, each computes 32x32
// (4 WMMAs per K-step). f32 tiles are register-pipelined (conversion
// to bf16 happens while staging to LDS).
// -------------------------------------------------------------------------
__global__ void proj_qkv_kernel(const float* __restrict__ X,
                                const float* __restrict__ W,
                                const float* __restrict__ bias,
                                bf16* __restrict__ out, float scale)
{
  __shared__ bf16 As[64][40];
  __shared__ bf16 Bs[128][40];

  const int tid    = threadIdx.x;
  const int lane   = tid & 31;
  const int wave   = tid >> 5;
  const int lane15 = lane & 15;
  const int half   = lane >> 4;

  const int m0 = blockIdx.x * 64;
  const int n0 = blockIdx.y * 128;
  const int wm = (wave & 1) * 32;
  const int wn = (wave >> 1) * 32;

  v8f acc00 = {}, acc01 = {}, acc10 = {}, acc11 = {};

  const int arow = tid >> 2;          // 0..63  (8 f32 per thread)
  const int acol = (tid & 3) * 8;
  const int brow = tid >> 1;          // 0..127 (16 f32 per thread)
  const int bcol = (tid & 1) * 16;

  const float* agp = X + (size_t)(m0 + arow) * ED + acol;
  const float* bgp = W + (size_t)(n0 + brow) * ED + bcol;

  float4 ar0 = *reinterpret_cast<const float4*>(agp + 0);
  float4 ar1 = *reinterpret_cast<const float4*>(agp + 4);
  float4 br0 = *reinterpret_cast<const float4*>(bgp + 0);
  float4 br1 = *reinterpret_cast<const float4*>(bgp + 4);
  float4 br2 = *reinterpret_cast<const float4*>(bgp + 8);
  float4 br3 = *reinterpret_cast<const float4*>(bgp + 12);

  for (int k0 = 0; k0 < ED; k0 += 32) {
    As[arow][acol+0]=tobf(ar0.x); As[arow][acol+1]=tobf(ar0.y);
    As[arow][acol+2]=tobf(ar0.z); As[arow][acol+3]=tobf(ar0.w);
    As[arow][acol+4]=tobf(ar1.x); As[arow][acol+5]=tobf(ar1.y);
    As[arow][acol+6]=tobf(ar1.z); As[arow][acol+7]=tobf(ar1.w);
    Bs[brow][bcol+ 0]=tobf(br0.x); Bs[brow][bcol+ 1]=tobf(br0.y);
    Bs[brow][bcol+ 2]=tobf(br0.z); Bs[brow][bcol+ 3]=tobf(br0.w);
    Bs[brow][bcol+ 4]=tobf(br1.x); Bs[brow][bcol+ 5]=tobf(br1.y);
    Bs[brow][bcol+ 6]=tobf(br1.z); Bs[brow][bcol+ 7]=tobf(br1.w);
    Bs[brow][bcol+ 8]=tobf(br2.x); Bs[brow][bcol+ 9]=tobf(br2.y);
    Bs[brow][bcol+10]=tobf(br2.z); Bs[brow][bcol+11]=tobf(br2.w);
    Bs[brow][bcol+12]=tobf(br3.x); Bs[brow][bcol+13]=tobf(br3.y);
    Bs[brow][bcol+14]=tobf(br3.z); Bs[brow][bcol+15]=tobf(br3.w);
    __syncthreads();

    if (k0 + 32 < ED) {
      const int kn = k0 + 32;
      ar0 = *reinterpret_cast<const float4*>(agp + kn + 0);
      ar1 = *reinterpret_cast<const float4*>(agp + kn + 4);
      br0 = *reinterpret_cast<const float4*>(bgp + kn + 0);
      br1 = *reinterpret_cast<const float4*>(bgp + kn + 4);
      br2 = *reinterpret_cast<const float4*>(bgp + kn + 8);
      br3 = *reinterpret_cast<const float4*>(bgp + kn + 12);
    }

    const int kOffA = half ? 8 : 0;
    const int kOffB = half ? 8 : 0;
    v16bf a0, a1, b0, b1;
    #pragma unroll
    for (int i = 0; i < 8; ++i) {
      a0[i]   = As[wm + lane15][kOffA + i];
      a0[8+i] = As[wm + lane15][kOffA + 16 + i];
      a1[i]   = As[wm + 16 + lane15][kOffA + i];
      a1[8+i] = As[wm + 16 + lane15][kOffA + 16 + i];
    }
    #pragma unroll
    for (int i = 0; i < 16; ++i) {
      b0[i] = Bs[wn + lane15][kOffB + i + (i & 8)];
      b1[i] = Bs[wn + 16 + lane15][kOffB + i + (i & 8)];
    }
    acc00 = wmma_bf16(a0, b0, acc00);
    acc01 = wmma_bf16(a0, b1, acc01);
    acc10 = wmma_bf16(a1, b0, acc10);
    acc11 = wmma_bf16(a1, b1, acc11);
    __syncthreads();
  }

  #pragma unroll
  for (int an = 0; an < 2; ++an) {
    const int gcol = n0 + wn + an*16 + lane15;
    const float bvv = bias[gcol];
    const int h = gcol >> 6;
    const int d = gcol & 63;
    #pragma unroll
    for (int r = 0; r < 8; ++r) {
      const float v0 = ((an ? acc01[r] : acc00[r]) + bvv) * scale;
      const float v1 = ((an ? acc11[r] : acc10[r]) + bvv) * scale;
      const int gm0  = m0 + wm + r + 8*half;
      const int gm1  = gm0 + 16;
      out[(((size_t)((gm0 >> 11)*NH + h)*NTOK + (gm0 & 2047)) * HD) + d] = tobf(v0);
      out[(((size_t)((gm1 >> 11)*NH + h)*NTOK + (gm1 & 2047)) * HD) + d] = tobf(v1);
    }
  }
}

// -------------------------------------------------------------------------
// Flash attention: grid (N/128, B*NH), 256 threads = 8 waves.
// Each wave owns 16 query rows; 64-key tiles. K tile is DMA'd to LDS with
// global_load_async_to_lds_b128 (ASYNCcnt), overlapping the V transpose
// done through VGPRs. 16 WMMAs per tile between 2 block barriers.
// -------------------------------------------------------------------------
__global__ void attn_kernel(const bf16* __restrict__ Qb,
                            const bf16* __restrict__ Kb,
                            const bf16* __restrict__ Vb,
                            bf16* __restrict__ Cb)
{
  __shared__ __align__(16) bf16 Ks[64][72];   // [key][d]    (144B rows, 16B aligned)
  __shared__ __align__(16) bf16 Vt[64][72];   // [d][key]    (transposed)
  __shared__ bf16 Ps[8][16][64];              // per-wave P scratch

  const int tid    = threadIdx.x;
  const int lane   = tid & 31;
  const int wave   = tid >> 5;
  const int lane15 = lane & 15;
  const int half   = lane >> 4;

  const int bh    = blockIdx.y;                  // b*NH + h
  const int qbase = blockIdx.x*128 + wave*16;

  const bf16* Qh = Qb + (size_t)bh * NTOK * HD;
  const bf16* Kh = Kb + (size_t)bh * NTOK * HD;
  const bf16* Vh = Vb + (size_t)bh * NTOK * HD;

  // resident Q fragments: two K=32 chunks covering d=0..63
  v16bf aq0, aq1;
  {
    const int qrow = qbase + lane15;
    const int kOff = half ? 8 : 0;
    const bf16* qp = Qh + (size_t)qrow * HD;
    #pragma unroll
    for (int i = 0; i < 8; ++i) {
      aq0[i]   = qp[kOff + i];
      aq0[8+i] = qp[kOff + 16 + i];
      aq1[i]   = qp[32 + kOff + i];
      aq1[8+i] = qp[32 + kOff + 16 + i];
    }
  }

  const v8f vzero = {};
  v8f acc[4];
  v8f s[4];
  #pragma unroll
  for (int c = 0; c < 4; ++c) acc[c] = vzero;

  float m_r[8], l_r[8];
  #pragma unroll
  for (int r = 0; r < 8; ++r) { m_r[r] = -1e30f; l_r[r] = 0.f; }

  const int krow = tid >> 2;          // 0..63 : key row, 16 bf16 per thread
  const int kdc  = (tid & 3) * 16;    // d base 0/16/32/48
  const unsigned ks_off = LDS_OFF(&Ks[krow][kdc]);

  for (int kt = 0; kt < NTOK; kt += 64) {
    // K tile: async DMA global -> LDS (2 x 16B per thread)
    {
      const bf16* kp = Kh + (size_t)(kt + krow)*HD + kdc;
      async_copy_b128(ks_off,      kp);
      async_copy_b128(ks_off + 16, kp + 8);
    }
    // V tile: through VGPRs, transposed into Vt (overlaps with K DMA)
    {
      const bf16* vp = Vh + (size_t)(kt + krow)*HD + kdc;
      uint4 v0v = *reinterpret_cast<const uint4*>(vp);
      uint4 v1v = *reinterpret_cast<const uint4*>(vp + 8);
      const bf16* vh0 = reinterpret_cast<const bf16*>(&v0v);
      const bf16* vh1 = reinterpret_cast<const bf16*>(&v1v);
      #pragma unroll
      for (int e = 0; e < 8; ++e) {
        Vt[kdc + e][krow]     = vh0[e];
        Vt[kdc + 8 + e][krow] = vh1[e];
      }
    }
    wait_async0();
    __syncthreads();

    // scores: 16 x 64 = four 16x16 C tiles, K=64 via two WMMAs each
    const int kOffB = half ? 8 : 0;
    #pragma unroll
    for (int t = 0; t < 4; ++t) {
      s[t] = vzero;
      v16bf blo, bhi;
      #pragma unroll
      for (int i = 0; i < 16; ++i) {
        blo[i] = Ks[t*16 + lane15][kOffB + i + (i & 8)];        // d 0..31
        bhi[i] = Ks[t*16 + lane15][32 + kOffB + i + (i & 8)];   // d 32..63
      }
      s[t] = wmma_bf16(aq0, blo, s[t]);
      s[t] = wmma_bf16(aq1, bhi, s[t]);
    }

    // online softmax (rows live on half-waves -> width-16 reductions)
    #pragma unroll
    for (int r = 0; r < 8; ++r) {
      const float a0 = s[0][r], a1 = s[1][r], a2 = s[2][r], a3 = s[3][r];
      float rm = fmaxf(fmaxf(a0, a1), fmaxf(a2, a3));
      #pragma unroll
      for (int off = 8; off >= 1; off >>= 1) rm = fmaxf(rm, __shfl_xor(rm, off, 16));
      const float mn = fmaxf(m_r[r], rm);
      const float c_ = __expf(m_r[r] - mn);
      const float p0 = __expf(a0 - mn);
      const float p1 = __expf(a1 - mn);
      const float p2 = __expf(a2 - mn);
      const float p3 = __expf(a3 - mn);
      float rs = (p0 + p1) + (p2 + p3);
      #pragma unroll
      for (int off = 8; off >= 1; off >>= 1) rs += __shfl_xor(rs, off, 16);
      l_r[r] = l_r[r]*c_ + rs;
      m_r[r] = mn;
      acc[0][r] *= c_; acc[1][r] *= c_; acc[2][r] *= c_; acc[3][r] *= c_;
      const int prow = r + 8*half;
      Ps[wave][prow][ 0 + lane15] = tobf(p0);
      Ps[wave][prow][16 + lane15] = tobf(p1);
      Ps[wave][prow][32 + lane15] = tobf(p2);
      Ps[wave][prow][48 + lane15] = tobf(p3);
    }
    // Ps is wave-private; DS ops are in-order within a wave. Pin ordering:
    __builtin_amdgcn_wave_barrier();

    // ctx += P(16x64) @ V(64x64) : two K=32 steps, four 16-wide d chunks
    {
      v16bf ap0, ap1;
      const int kOffA = half ? 8 : 0;
      #pragma unroll
      for (int i = 0; i < 8; ++i) {
        ap0[i]   = Ps[wave][lane15][kOffA + i];
        ap0[8+i] = Ps[wave][lane15][kOffA + 16 + i];
        ap1[i]   = Ps[wave][lane15][32 + kOffA + i];
        ap1[8+i] = Ps[wave][lane15][32 + kOffA + 16 + i];
      }
      #pragma unroll
      for (int c = 0; c < 4; ++c) {
        const int dcol = c*16 + lane15;
        v16bf bv0, bv1;
        #pragma unroll
        for (int i = 0; i < 16; ++i) {
          bv0[i] = Vt[dcol][kOffB + i + (i & 8)];        // keys 0..31
          bv1[i] = Vt[dcol][32 + kOffB + i + (i & 8)];   // keys 32..63
        }
        acc[c] = wmma_bf16(ap0, bv0, acc[c]);
        acc[c] = wmma_bf16(ap1, bv1, acc[c]);
      }
    }
    __syncthreads();
  }

  // normalize and store ctx bf16 as [B, N, NH, HD]
  const int bidx = bh >> 4;
  const int h    = bh & 15;
  #pragma unroll
  for (int r = 0; r < 8; ++r) {
    const float inv = 1.0f / l_r[r];
    const int qrow  = qbase + r + 8*half;
    bf16* op = Cb + (((size_t)(bidx*NTOK + qrow) * NH + h) * HD);
    op[ 0 + lane15] = tobf(acc[0][r] * inv);
    op[16 + lane15] = tobf(acc[1][r] * inv);
    op[32 + lane15] = tobf(acc[2][r] * inv);
    op[48 + lane15] = tobf(acc[3][r] * inv);
  }
}

// -------------------------------------------------------------------------
// Output projection: out = ctx @ Wo^T + bo  (ctx bf16 [4096,1024], out f32)
// A tile (already bf16) is DMA'd with async LDS copies; B tile (f32)
// keeps the register pipeline for conversion.
// -------------------------------------------------------------------------
__global__ void out_proj_kernel(const bf16* __restrict__ Cx,
                                const float* __restrict__ W,
                                const float* __restrict__ bias,
                                float* __restrict__ out)
{
  __shared__ __align__(16) bf16 As[64][48];   // 96B rows, 16B aligned
  __shared__ bf16 Bs[128][40];

  const int tid    = threadIdx.x;
  const int lane   = tid & 31;
  const int wave   = tid >> 5;
  const int lane15 = lane & 15;
  const int half   = lane >> 4;

  const int m0 = blockIdx.x * 64;
  const int n0 = blockIdx.y * 128;
  const int wm = (wave & 1) * 32;
  const int wn = (wave >> 1) * 32;

  v8f acc00 = {}, acc01 = {}, acc10 = {}, acc11 = {};

  const int arow = tid >> 2;          // 0..63  (8 bf16 = 16B per thread)
  const int acol = (tid & 3) * 8;
  const int brow = tid >> 1;          // 0..127 (16 f32 per thread)
  const int bcol = (tid & 1) * 16;

  const bf16*  agp = Cx + (size_t)(m0 + arow) * ED + acol;
  const float* bgp = W  + (size_t)(n0 + brow) * ED + bcol;
  const unsigned as_off = LDS_OFF(&As[arow][acol]);

  float4 br0 = *reinterpret_cast<const float4*>(bgp + 0);
  float4 br1 = *reinterpret_cast<const float4*>(bgp + 4);
  float4 br2 = *reinterpret_cast<const float4*>(bgp + 8);
  float4 br3 = *reinterpret_cast<const float4*>(bgp + 12);

  for (int k0 = 0; k0 < ED; k0 += 32) {
    // A tile: async DMA global -> LDS (16B per thread)
    async_copy_b128(as_off, agp + k0);
    // B tile: staged registers, f32 -> bf16
    Bs[brow][bcol+ 0]=tobf(br0.x); Bs[brow][bcol+ 1]=tobf(br0.y);
    Bs[brow][bcol+ 2]=tobf(br0.z); Bs[brow][bcol+ 3]=tobf(br0.w);
    Bs[brow][bcol+ 4]=tobf(br1.x); Bs[brow][bcol+ 5]=tobf(br1.y);
    Bs[brow][bcol+ 6]=tobf(br1.z); Bs[brow][bcol+ 7]=tobf(br1.w);
    Bs[brow][bcol+ 8]=tobf(br2.x); Bs[brow][bcol+ 9]=tobf(br2.y);
    Bs[brow][bcol+10]=tobf(br2.z); Bs[brow][bcol+11]=tobf(br2.w);
    Bs[brow][bcol+12]=tobf(br3.x); Bs[brow][bcol+13]=tobf(br3.y);
    Bs[brow][bcol+14]=tobf(br3.z); Bs[brow][bcol+15]=tobf(br3.w);
    wait_async0();
    __syncthreads();

    if (k0 + 32 < ED) {
      const int kn = k0 + 32;
      br0 = *reinterpret_cast<const float4*>(bgp + kn + 0);
      br1 = *reinterpret_cast<const float4*>(bgp + kn + 4);
      br2 = *reinterpret_cast<const float4*>(bgp + kn + 8);
      br3 = *reinterpret_cast<const float4*>(bgp + kn + 12);
    }

    const int kOffA = half ? 8 : 0;
    const int kOffB = half ? 8 : 0;
    v16bf a0, a1, b0, b1;
    #pragma unroll
    for (int i = 0; i < 8; ++i) {
      a0[i]   = As[wm + lane15][kOffA + i];
      a0[8+i] = As[wm + lane15][kOffA + 16 + i];
      a1[i]   = As[wm + 16 + lane15][kOffA + i];
      a1[8+i] = As[wm + 16 + lane15][kOffA + 16 + i];
    }
    #pragma unroll
    for (int i = 0; i < 16; ++i) {
      b0[i] = Bs[wn + lane15][kOffB + i + (i & 8)];
      b1[i] = Bs[wn + 16 + lane15][kOffB + i + (i & 8)];
    }
    acc00 = wmma_bf16(a0, b0, acc00);
    acc01 = wmma_bf16(a0, b1, acc01);
    acc10 = wmma_bf16(a1, b0, acc10);
    acc11 = wmma_bf16(a1, b1, acc11);
    __syncthreads();
  }

  #pragma unroll
  for (int an = 0; an < 2; ++an) {
    const int gcol = n0 + wn + an*16 + lane15;
    const float bvv = bias[gcol];
    #pragma unroll
    for (int r = 0; r < 8; ++r) {
      const int gm0 = m0 + wm + r + 8*half;
      const int gm1 = gm0 + 16;
      out[(size_t)gm0 * ED + gcol] = (an ? acc01[r] : acc00[r]) + bvv;
      out[(size_t)gm1 * ED + gcol] = (an ? acc11[r] : acc10[r]) + bvv;
    }
  }
}

// -------------------------------------------------------------------------
extern "C" void kernel_launch(void* const* d_in, const int* in_sizes, int n_in,
                              void* d_out, int out_size, void* d_ws, size_t ws_size,
                              hipStream_t stream) {
  const float* x  = (const float*)d_in[0];
  const float* wq = (const float*)d_in[1];
  const float* bq = (const float*)d_in[2];
  const float* wk = (const float*)d_in[3];
  const float* bk = (const float*)d_in[4];
  const float* wv = (const float*)d_in[5];
  const float* bv = (const float*)d_in[6];
  const float* wo = (const float*)d_in[7];
  const float* bo = (const float*)d_in[8];

  const size_t perBuf = (size_t)BATCH * NH * NTOK * HD;  // 4M bf16 elems each
  bf16* Qb = (bf16*)d_ws;
  bf16* Kb = Qb + perBuf;
  bf16* Vb = Kb + perBuf;
  bf16* Cb = Vb + perBuf;

  dim3 gproj(MTOT / 64, ED / 128);
  proj_qkv_kernel<<<gproj, 256, 0, stream>>>(x, wq, bq, Qb, 0.125f); // 1/sqrt(64)
  proj_qkv_kernel<<<gproj, 256, 0, stream>>>(x, wk, bk, Kb, 1.0f);
  proj_qkv_kernel<<<gproj, 256, 0, stream>>>(x, wv, bv, Vb, 1.0f);

  attn_kernel<<<dim3(NTOK / 128, BATCH * NH), 256, 0, stream>>>(Qb, Kb, Vb, Cb);

  out_proj_kernel<<<gproj, 256, 0, stream>>>(Cb, wo, bo, (float*)d_out);
}
